// Encoder_46188078301477
// MI455X (gfx1250) — matrix operand order repaired and checked by reading, not compiled, not run on hardware
//
#include <hip/hip_runtime.h>
#include <hip/hip_bf16.h>
#include <math.h>
#include <stdint.h>

#define L_ 6
#define C_ 192
#define FC_ 768
#define H_ 2
#define KC_ 96
#define B_ 4
#define T_ 2048
#define BT_ (B_*T_)
#define KW_ 3
#define W_ 4
#define TP_ (T_+2)
#define EPS_ 1e-5f

typedef __attribute__((ext_vector_type(16))) __bf16 v16bf;
typedef __attribute__((ext_vector_type(8)))  float  v8f;
typedef __attribute__((ext_vector_type(4)))  unsigned int u32x4;
typedef __attribute__((ext_vector_type(4)))  unsigned int tdm_u4;
typedef __attribute__((ext_vector_type(8)))  int          tdm_i8;
typedef __attribute__((ext_vector_type(4)))  int          tdm_i4;

union ABFrag { u32x4 u[2]; v16bf v; };

__device__ __forceinline__ unsigned short f2bf(float f) {
  unsigned int u = __float_as_uint(f);
  u += 0x7FFFu + ((u >> 16) & 1u);         // round-to-nearest-even
  return (unsigned short)(u >> 16);
}

__device__ __forceinline__ v8f wmma_bf16(v16bf a, v16bf b, v8f c) {
  // (neg_a, A, neg_b, B, c_mod, C, reuse_a, reuse_b)
  return __builtin_amdgcn_wmma_f32_16x16x32_bf16(false, a, false, b, (short)0, c,
                                                 false, false);
}

// Issue a TDM 2D tile load: rows x width(bf16 elems), row stride strideElems,
// global src -> LDS offset ldsOff. Must be executed by one full wave.
__device__ __forceinline__ void tdm_load_2d(const unsigned short* src,
                                            unsigned ldsOff, int width,
                                            int rows, int strideElems) {
  uint64_t ga = (uint64_t)(uintptr_t)src;
  tdm_u4 g0;
  g0[0] = 1u;                                 // count=1, user descriptor
  g0[1] = ldsOff;                             // lds_addr (bytes)
  g0[2] = (unsigned)ga;                       // global_addr[31:0]
  g0[3] = (unsigned)(ga >> 32) | (2u << 30);  // global_addr[56:32] | type=2
  tdm_i8 g1;
  unsigned w = (unsigned)width, st = (unsigned)strideElems, r = (unsigned)rows;
  g1[0] = (int)(1u << 16);                    // data_size = 1 -> 2 bytes
  g1[1] = (int)((w & 0xFFFFu) << 16);         // tensor_dim0[15:0] @ bits 63:48
  g1[2] = (int)((w >> 16) | (r << 16));       // tensor_dim0[31:16] | tensor_dim1[15:0]
  g1[3] = (int)(w << 16);                     // tensor_dim1[31:16]=0 | tile_dim0
  g1[4] = (int)r;                             // tile_dim1 | tile_dim2=0
  g1[5] = (int)st;                            // tensor_dim0_stride[31:0]
  g1[6] = 0;                                  // stride0[47:32] | stride1[15:0]
  g1[7] = 0;
  tdm_i4 g2 = {0,0,0,0};
  tdm_i4 g3 = {0,0,0,0};
#if __has_include(<hip/amd_detail/amd_gfx1250_TDM.h>)
  tdm_i8 g4 = {0,0,0,0,0,0,0,0};
  __builtin_amdgcn_tensor_load_to_lds(g0, g1, g2, g3, g4, 0);
#else
  __builtin_amdgcn_tensor_load_to_lds(g0, g1, g2, g3, 0);
#endif
  __builtin_amdgcn_s_wait_tensorcnt(0);
}

// ---------------------------------------------------------------------------
// Cooperative GEMM: block = 8 waves sharing one 16-out-channel strip.
// A panel (16 x Cin bf16) DMA'd to LDS by the TDM once per tap; each wave
// computes a 16(M) x 64(N) tile block (A-fragment reused across 4 WMMAs).
//   Y[n, o] = act( scale * (sum_dk sum_c W[dk][o][c] * X[row(n,dk)][c] + bias[o]) )
// modes: 0 -> f32 [BT][Cout], 1 -> bf16 [BT][Cout],
//        2 -> bf16 channel-major [B][Cout][T], 3 -> bf16 padded [B][T+2][Cout]
// paddedIn: X rows indexed as b*(T+2) + t + dk (zero pad rows at t=-1, t=T)
// ---------------------------------------------------------------------------
__global__ __launch_bounds__(256) void gemm16(
    const unsigned short* __restrict__ Wt, const unsigned short* __restrict__ X,
    const float* __restrict__ bias, float* __restrict__ outF,
    unsigned short* __restrict__ outB,
    int Cin, int Cout, int kTaps, int tapStride, int paddedIn,
    float scaleOut, int doRelu, int mode)
{
  __shared__ unsigned short ldsW[16 * FC_];   // up to 16 x 768 bf16 = 24 KB

  const int wave = threadIdx.x >> 5;
  const int lane = threadIdx.x & 31;
  const int nI   = lane & 15;
  const int half = lane >> 4;

  const int nBlocks = BT_ / 512;              // 16 super-blocks of 512 tokens
  const int blkN = blockIdx.x % nBlocks;
  const int tO   = blockIdx.x / nBlocks;
  const int o0   = tO << 4;
  const int nBase = blkN * 512 + wave * 64;   // 4 token tiles per wave
  const int b  = nBase / T_;
  const int t0 = nBase - b * T_;

  v8f acc[4];
#pragma unroll
  for (int j = 0; j < 4; ++j) acc[j] = (v8f){0.f,0.f,0.f,0.f,0.f,0.f,0.f,0.f};

  const unsigned ldsOff = (unsigned)(uintptr_t)(void*)&ldsW[0];

  for (int dk = 0; dk < kTaps; ++dk) {
    __syncthreads();                          // prior tap's LDS reads done
    if (wave == 0)
      tdm_load_2d(Wt + (size_t)dk * tapStride + (size_t)o0 * Cin,
                  ldsOff, Cin, 16, Cin);
    __syncthreads();                          // panel visible to all waves

    const unsigned short* xB = paddedIn
        ? X + (size_t)(b * TP_ + t0 + dk) * Cin
        : X + (size_t)nBase * Cin;
    for (int c0 = 0; c0 < Cin; c0 += 32) {
      ABFrag a;
      a.u[0] = *(const u32x4*)(&ldsW[nI * Cin + c0 + half * 8]);
      a.u[1] = *(const u32x4*)(&ldsW[nI * Cin + c0 + 16 + half * 8]);
#pragma unroll
      for (int j = 0; j < 4; ++j) {
        const unsigned short* xR = xB + (size_t)(16*j + nI) * Cin;
        ABFrag bb;
        bb.u[0] = *(const u32x4*)(xR + c0 + half * 8);
        bb.u[1] = *(const u32x4*)(xR + c0 + 16 + half * 8);
        acc[j] = wmma_bf16(a.v, bb.v, acc[j]);
      }
    }
  }

  const int ob = o0 + half * 8;               // 8 consecutive out channels
#pragma unroll
  for (int j = 0; j < 4; ++j) {
    const int tok = nBase + 16*j + nI;
    const int tt  = t0 + 16*j + nI;
#pragma unroll
    for (int r = 0; r < 8; ++r) {
      float v = (acc[j][r] + bias[ob + r]) * scaleOut;
      if (doRelu) v = fmaxf(v, 0.f);
      if (mode == 0)      outF[(size_t)tok * Cout + ob + r] = v;
      else if (mode == 1) outB[(size_t)tok * Cout + ob + r] = f2bf(v);
      else if (mode == 2) outB[((size_t)b * Cout + ob + r) * T_ + tt] = f2bf(v);
      else                outB[(size_t)(b * TP_ + tt + 1) * Cout + ob + r] = f2bf(v);
    }
  }
}

// ---------------------------------------------------------------------------
// Relative-key logits: RL[b,h,t,rp] = sum_kc q[b,h,t,kc] * erk_pad[rp,kc]
// A = erk_pad (16 rows, rp 9..15 zero), B = q^T.  One wave per 16-token tile.
// ---------------------------------------------------------------------------
__global__ __launch_bounds__(256) void relk16(
    const unsigned short* __restrict__ Q, const unsigned short* __restrict__ erkb,
    float* __restrict__ RL)
{
  const int wid  = blockIdx.x * 8 + (threadIdx.x >> 5);
  const int lane = threadIdx.x & 31;
  const int nI = lane & 15, half = lane >> 4;
  const int tilesT = T_ >> 4;
  const int tq = (wid % tilesT) << 4;
  const int bh = wid / tilesT;
  const int h = bh % H_, b = bh / H_;

  v8f acc = {0.f,0.f,0.f,0.f,0.f,0.f,0.f,0.f};
  const unsigned short* eR = erkb + nI * KC_;
  const unsigned short* qR = Q + (size_t)(b * T_ + tq + nI) * C_ + h * KC_;
#pragma unroll
  for (int c0 = 0; c0 < KC_; c0 += 32) {
    ABFrag a, bb;
    a.u[0]  = *(const u32x4*)(eR + c0 + half * 8);
    a.u[1]  = *(const u32x4*)(eR + c0 + 16 + half * 8);
    bb.u[0] = *(const u32x4*)(qR + c0 + half * 8);
    bb.u[1] = *(const u32x4*)(qR + c0 + 16 + half * 8);
    acc = wmma_bf16(a.v, bb.v, acc);
  }
  float* o = RL + ((size_t)bh * T_ + tq + nI) * 16 + half * 8;
#pragma unroll
  for (int r = 0; r < 8; ++r) o[r] = acc[r];
}

// ---------------------------------------------------------------------------
// Flash attention + banded relative terms. One wave (=one block) per
// (b, h, 16-query tile); s-blocks of 32 (two score tiles -> one 16x32 P frag).
// ---------------------------------------------------------------------------
__global__ __launch_bounds__(32) void attn_flash(
    const unsigned short* __restrict__ Q, const unsigned short* __restrict__ Kb,
    const unsigned short* __restrict__ Vch, const float* __restrict__ RL,
    const float* __restrict__ erv, unsigned short* __restrict__ Ot)
{
  const int wid = blockIdx.x;
  const int tilesT = T_ >> 4;
  const int tq = (wid % tilesT) << 4;
  const int bh = wid / tilesT;
  const int h = bh % H_, b = bh / H_;
  const int lane = threadIdx.x;
  const int nI = lane & 15, half = lane >> 4;

  __shared__ float ervS[(2*W_+1) * KC_];
  __shared__ unsigned short pT[16 * 32];
  __shared__ float pb[16 * 16];

  for (int i = lane; i < (2*W_+1)*KC_; i += 32) ervS[i] = erv[i];
  for (int i = lane; i < 256; i += 32) pb[i] = 0.f;
  __syncthreads();

  // resident q A-fragments (16 x 96 bf16 = 3 chunks)
  v16bf qa[3];
  {
    const unsigned short* qR = Q + (size_t)(b*T_ + tq + nI) * C_ + h * KC_;
#pragma unroll
    for (int kc = 0; kc < 3; ++kc) {
      ABFrag a;
      a.u[0] = *(const u32x4*)(qR + kc*32 + half*8);
      a.u[1] = *(const u32x4*)(qR + kc*32 + 16 + half*8);
      qa[kc] = a.v;
    }
  }

  v8f outAcc[6];
#pragma unroll
  for (int j = 0; j < 6; ++j) outAcc[j] = (v8f){0.f,0.f,0.f,0.f,0.f,0.f,0.f,0.f};
  float mRow[8], lRow[8];
#pragma unroll
  for (int r = 0; r < 8; ++r) { mRow[r] = -1e30f; lRow[r] = 0.f; }

  const float* rlB = RL + ((size_t)bh * T_ + tq) * 16;
  const unsigned short* kBase = Kb  + ((size_t)b * T_) * C_ + h * KC_;
  const unsigned short* vBase = Vch + ((size_t)b * C_ + h * KC_) * T_;

  auto computeS = [&](int s0, v8f& S0, v8f& S1) {
    S0 = (v8f){0.f,0.f,0.f,0.f,0.f,0.f,0.f,0.f};
    S1 = (v8f){0.f,0.f,0.f,0.f,0.f,0.f,0.f,0.f};
    const unsigned short* k0 = kBase + (size_t)(s0 + nI) * C_;
    const unsigned short* k1 = k0 + (size_t)16 * C_;
#pragma unroll
    for (int kc = 0; kc < 3; ++kc) {
      ABFrag b0, b1;
      b0.u[0] = *(const u32x4*)(k0 + kc*32 + half*8);
      b0.u[1] = *(const u32x4*)(k0 + kc*32 + 16 + half*8);
      b1.u[0] = *(const u32x4*)(k1 + kc*32 + half*8);
      b1.u[1] = *(const u32x4*)(k1 + kc*32 + 16 + half*8);
      S0 = wmma_bf16(qa[kc], b0.v, S0);
      S1 = wmma_bf16(qa[kc], b1.v, S1);
    }
    if (s0 + 31 >= tq - W_ && s0 <= tq + 15 + W_) {   // banded rel-k logits
#pragma unroll
      for (int r = 0; r < 8; ++r) {
        int m = half*8 + r;
        int rp0 = s0 + nI - (tq + m) + W_;
        if (rp0 >= 0 && rp0 <= 2*W_) S0[r] += rlB[m*16 + rp0];
        int rp1 = rp0 + 16;
        if (rp1 >= 0 && rp1 <= 2*W_) S1[r] += rlB[m*16 + rp1];
      }
    }
  };

  for (int s0 = 0; s0 < T_; s0 += 32) {
    v8f S0, S1;
    computeS(s0, S0, S1);

    // online softmax (rows m = half*8 + r live on the 16 lanes of this half)
#pragma unroll
    for (int r = 0; r < 8; ++r) {
      float mx = fmaxf(S0[r], S1[r]);
#pragma unroll
      for (int d = 1; d < 16; d <<= 1) mx = fmaxf(mx, __shfl_xor(mx, d, 32));
      float mNew = fmaxf(mRow[r], mx);
      float sc = __expf(mRow[r] - mNew);
      float p0 = __expf(S0[r] - mNew);
      float p1 = __expf(S1[r] - mNew);
      float ls = p0 + p1;
#pragma unroll
      for (int d = 1; d < 16; d <<= 1) ls += __shfl_xor(ls, d, 32);
      lRow[r] = lRow[r] * sc + ls;
      mRow[r] = mNew;
      S0[r] = p0; S1[r] = p1;
#pragma unroll
      for (int j = 0; j < 6; ++j) outAcc[j][r] *= sc;
    }

    // C/D layout -> A layout transpose for P via LDS
#pragma unroll
    for (int r = 0; r < 8; ++r) {
      int m = half*8 + r;
      pT[m*32 + nI]      = f2bf(S0[r]);
      pT[m*32 + 16 + nI] = f2bf(S1[r]);
    }
    __syncthreads();
    ABFrag pa;
    pa.u[0] = *(const u32x4*)(&pT[nI*32 + half*8]);
    pa.u[1] = *(const u32x4*)(&pT[nI*32 + 16 + half*8]);
    __syncthreads();

    // O += P x V  (V channel-major -> contiguous B-fragment loads)
#pragma unroll
    for (int j = 0; j < 6; ++j) {
      const unsigned short* vR = vBase + (size_t)(16*j + nI) * T_ + s0;
      ABFrag vb;
      vb.u[0] = *(const u32x4*)(vR + half*8);
      vb.u[1] = *(const u32x4*)(vR + 16 + half*8);
      outAcc[j] = wmma_bf16(pa.v, vb.v, outAcc[j]);
    }
  }

  float invl[8];
#pragma unroll
  for (int r = 0; r < 8; ++r) invl[r] = 1.f / lRow[r];

  // recompute band s-blocks to harvest normalized band probabilities pb[m][rp]
  int sLo = tq - W_; if (sLo < 0) sLo = 0;
  sLo = (sLo / 32) * 32;
  for (int s0 = sLo; s0 < T_ && s0 <= tq + 15 + W_; s0 += 32) {
    v8f S0, S1;
    computeS(s0, S0, S1);
#pragma unroll
    for (int r = 0; r < 8; ++r) {
      int m = half*8 + r;
      int rp0 = s0 + nI - (tq + m) + W_;
      if (rp0 >= 0 && rp0 <= 2*W_)
        pb[m*16 + rp0] = __expf(S0[r] - mRow[r]) * invl[r];
      int rp1 = rp0 + 16;
      if (rp1 >= 0 && rp1 <= 2*W_)
        pb[m*16 + rp1] = __expf(S1[r] - mRow[r]) * invl[r];
    }
  }
  __syncthreads();

  // epilogue: out = acc/l + sum_rp pb[m][rp] * erv[rp][d]
  unsigned short* oB = Ot + (size_t)(b*T_ + tq) * C_ + h * KC_;
#pragma unroll
  for (int r = 0; r < 8; ++r) {
    int m = half*8 + r;
    float pbr[2*W_+1];
#pragma unroll
    for (int rp = 0; rp <= 2*W_; ++rp) pbr[rp] = pb[m*16 + rp];
#pragma unroll
    for (int j = 0; j < 6; ++j) {
      int d = 16*j + nI;
      float delta = 0.f;
#pragma unroll
      for (int rp = 0; rp <= 2*W_; ++rp) delta += pbr[rp] * ervS[rp*KC_ + d];
      float val = outAcc[j][r] * invl[r] + delta;
      oB[(size_t)m * C_ + d] = f2bf(val);
    }
  }
}

// ---------------------------------------------------------------------------
// Residual-add + channel LayerNorm; writes f32 master + bf16 copy
// (padMode: bf16 goes to the zero-padded [B][T+2][C] conv-input buffer)
// ---------------------------------------------------------------------------
__global__ __launch_bounds__(256) void ln_fuse(
    const float* __restrict__ xin, const float* __restrict__ res,
    const float* __restrict__ g, const float* __restrict__ be,
    float* __restrict__ xout, unsigned short* __restrict__ bout, int padMode)
{
  const int tok  = blockIdx.x * 8 + (threadIdx.x >> 5);
  const int lane = threadIdx.x & 31;
  const float* xr = xin + (size_t)tok * C_;
  const float* rr = res + (size_t)tok * C_;
  float v[6]; float s = 0.f;
#pragma unroll
  for (int i = 0; i < 6; ++i) { v[i] = xr[lane + 32*i] + rr[lane + 32*i]; s += v[i]; }
#pragma unroll
  for (int d = 1; d < 32; d <<= 1) s += __shfl_xor(s, d, 32);
  float mean = s * (1.f / C_);
  float vs = 0.f;
#pragma unroll
  for (int i = 0; i < 6; ++i) { float d0 = v[i] - mean; vs += d0*d0; }
#pragma unroll
  for (int d = 1; d < 32; d <<= 1) vs += __shfl_xor(vs, d, 32);
  float rstd = rsqrtf(vs * (1.f / C_) + EPS_);
  const int b = tok / T_, t = tok - b * T_;
  float* xo = xout + (size_t)tok * C_;
  unsigned short* bo = padMode ? bout + (size_t)(b * TP_ + t + 1) * C_
                               : bout + (size_t)tok * C_;
#pragma unroll
  for (int i = 0; i < 6; ++i) {
    int c = lane + 32*i;
    float y = (v[i] - mean) * rstd * g[c] + be[c];
    xo[c] = y;
    bo[c] = f2bf(y);
  }
}

// --------------------------- data movement helpers -------------------------
__global__ void reshape_in(const float* __restrict__ x, const float* __restrict__ mask,
                           float* __restrict__ xm, unsigned short* __restrict__ xt)
{
  int idx = blockIdx.x * 256 + threadIdx.x;
  if (idx >= B_*C_*T_) return;
  int t = idx % T_; int c = (idx / T_) % C_; int b = idx / (T_*C_);
  float v = x[idx] * mask[b*T_ + t];
  size_t o = (size_t)(b*T_ + t) * C_ + c;
  xm[o] = v; xt[o] = f2bf(v);
}

__global__ void writeout(const float* __restrict__ xm, const float* __restrict__ mask,
                         float* __restrict__ out)
{
  int idx = blockIdx.x * 256 + threadIdx.x;
  if (idx >= B_*C_*T_) return;
  int t = idx % T_; int c = (idx / T_) % C_; int b = idx / (T_*C_);
  out[idx] = xm[(size_t)(b*T_ + t) * C_ + c] * mask[b*T_ + t];
}

__global__ void wcast(const float* __restrict__ in, unsigned short* __restrict__ out, int n) {
  int i = blockIdx.x * 256 + threadIdx.x;
  if (i < n) out[i] = f2bf(in[i]);
}

// [L][O][Ci][K] f32 -> [L][K][O][Ci] bf16
__global__ void wcast_conv(const float* __restrict__ in, unsigned short* __restrict__ out,
                           int O, int Ci, int n) {
  int i = blockIdx.x * 256 + threadIdx.x;
  if (i >= n) return;
  int dk = i % KW_;
  int c  = (i / KW_) % Ci;
  int o  = (i / (KW_ * Ci)) % O;
  int l  = i / (KW_ * Ci * O);
  out[(((size_t)l * KW_ + dk) * O + o) * Ci + c] = f2bf(in[i]);
}

// [L][9][KC] f32 -> [L][16][KC] bf16 (rows 9..15 zero)
__global__ void wcast_rel(const float* __restrict__ in, unsigned short* __restrict__ out, int n) {
  int i = blockIdx.x * 256 + threadIdx.x;
  if (i >= n) return;
  int c  = i % KC_;
  int rp = (i / KC_) % 16;
  int l  = i / (16 * KC_);
  out[i] = (rp < (2*W_+1)) ? f2bf(in[((size_t)l * (2*W_+1) + rp) * KC_ + c])
                           : (unsigned short)0;
}

// ---------------------------------------------------------------------------
extern "C" void kernel_launch(void* const* d_in, const int* in_sizes, int n_in,
                              void* d_out, int out_size, void* d_ws, size_t ws_size,
                              hipStream_t stream) {
  (void)in_sizes; (void)n_in; (void)out_size; (void)ws_size;
  const float* x    = (const float*)d_in[0];
  const float* mask = (const float*)d_in[1];
  const float* Wq = (const float*)d_in[2];  const float* bq = (const float*)d_in[3];
  const float* Wk = (const float*)d_in[4];  const float* bk = (const float*)d_in[5];
  const float* Wv = (const float*)d_in[6];  const float* bv = (const float*)d_in[7];
  const float* Wo = (const float*)d_in[8];  const float* bo = (const float*)d_in[9];
  const float* erk = (const float*)d_in[10]; const float* erv = (const float*)d_in[11];
  const float* g1 = (const float*)d_in[12]; const float* be1 = (const float*)d_in[13];
  const float* fw1 = (const float*)d_in[14]; const float* fb1 = (const float*)d_in[15];
  const float* fw2 = (const float*)d_in[16]; const float* fb2 = (const float*)d_in[17];
  const float* g2 = (const float*)d_in[18]; const float* be2 = (const float*)d_in[19];

  char* p = (char*)d_ws;
  auto take = [&](size_t bytes) { char* q = p; p += (bytes + 255) & ~(size_t)255; return q; };

  float*          xm   = (float*)take((size_t)BT_*C_*4);
  unsigned short* xt   = (unsigned short*)take((size_t)BT_*C_*2);
  unsigned short* qb   = (unsigned short*)take((size_t)BT_*C_*2);
  unsigned short* kb   = (unsigned short*)take((size_t)BT_*C_*2);
  unsigned short* vch  = (unsigned short*)take((size_t)BT_*C_*2);
  unsigned short* ot   = (unsigned short*)take((size_t)BT_*C_*2);
  float*          yo   = (float*)take((size_t)BT_*C_*4);
  float*          rl   = (float*)take((size_t)B_*H_*T_*16*4);
  size_t xpadBytes = (size_t)B_*TP_*C_*2;
  size_t hpadBytes = (size_t)B_*TP_*FC_*2;
  unsigned short* xpad = (unsigned short*)take(xpadBytes);
  unsigned short* hpad = (unsigned short*)take(hpadBytes);
  unsigned short* wqb  = (unsigned short*)take((size_t)L_*C_*C_*2);
  unsigned short* wkb  = (unsigned short*)take((size_t)L_*C_*C_*2);
  unsigned short* wvb  = (unsigned short*)take((size_t)L_*C_*C_*2);
  unsigned short* wob  = (unsigned short*)take((size_t)L_*C_*C_*2);
  unsigned short* wc1b = (unsigned short*)take((size_t)L_*KW_*FC_*C_*2);
  unsigned short* wc2b = (unsigned short*)take((size_t)L_*KW_*C_*FC_*2);
  unsigned short* erkb = (unsigned short*)take((size_t)L_*16*KC_*2);

  const float scale = 1.0f / sqrtf((float)KC_);

  // weight / embedding conversions (deterministic, every call)
  int nW = L_*C_*C_;
  wcast<<<(nW+255)/256,256,0,stream>>>(Wq, wqb, nW);
  wcast<<<(nW+255)/256,256,0,stream>>>(Wk, wkb, nW);
  wcast<<<(nW+255)/256,256,0,stream>>>(Wv, wvb, nW);
  wcast<<<(nW+255)/256,256,0,stream>>>(Wo, wob, nW);
  int nC1 = L_*FC_*C_*KW_;
  wcast_conv<<<(nC1+255)/256,256,0,stream>>>(fw1, wc1b, FC_, C_, nC1);
  int nC2 = L_*C_*FC_*KW_;
  wcast_conv<<<(nC2+255)/256,256,0,stream>>>(fw2, wc2b, C_, FC_, nC2);
  int nR = L_*16*KC_;
  wcast_rel<<<(nR+255)/256,256,0,stream>>>(erk, erkb, nR);

  hipMemsetAsync(xpad, 0, xpadBytes, stream);
  hipMemsetAsync(hpad, 0, hpadBytes, stream);

  int nE = B_*C_*T_;
  reshape_in<<<(nE+255)/256,256,0,stream>>>(x, mask, xm, xt);

  const int nSuper      = BT_ / 512;                 // 16
  const int projBlocks  = (C_/16)  * nSuper;         // 192
  const int conv1Blocks = (FC_/16) * nSuper;         // 768
  const int attnBlocks  = B_ * H_ * (T_/16);         // 1024
  const int relkBlocks  = attnBlocks / 8;            // 128
  const int lnBlocks    = BT_ / 8;                   // 1024

  for (int i = 0; i < L_; ++i) {
    const unsigned short* wq_i = wqb + (size_t)i*C_*C_;
    const unsigned short* wk_i = wkb + (size_t)i*C_*C_;
    const unsigned short* wv_i = wvb + (size_t)i*C_*C_;
    const unsigned short* wo_i = wob + (size_t)i*C_*C_;
    const unsigned short* w1_i = wc1b + (size_t)i*KW_*FC_*C_;
    const unsigned short* w2_i = wc2b + (size_t)i*KW_*C_*FC_;

    gemm16<<<projBlocks,256,0,stream>>>(wq_i, xt, bq + i*C_, nullptr, qb,
                                        C_, C_, 1, 0, 0, scale, 0, 1);
    gemm16<<<projBlocks,256,0,stream>>>(wk_i, xt, bk + i*C_, nullptr, kb,
                                        C_, C_, 1, 0, 0, 1.f, 0, 1);
    gemm16<<<projBlocks,256,0,stream>>>(wv_i, xt, bv + i*C_, nullptr, vch,
                                        C_, C_, 1, 0, 0, 1.f, 0, 2);
    relk16<<<relkBlocks,256,0,stream>>>(qb, erkb + (size_t)i*16*KC_, rl);
    attn_flash<<<attnBlocks,32,0,stream>>>(qb, kb, vch, rl,
                                           erv + (size_t)i*(2*W_+1)*KC_, ot);
    gemm16<<<projBlocks,256,0,stream>>>(wo_i, ot, bo + i*C_, yo, nullptr,
                                        C_, C_, 1, 0, 0, 1.f, 0, 0);
    ln_fuse<<<lnBlocks,256,0,stream>>>(xm, yo, g1 + i*C_, be1 + i*C_, xm, xpad, 1);
    gemm16<<<conv1Blocks,256,0,stream>>>(w1_i, xpad, fb1 + i*FC_, nullptr, hpad,
                                         C_, FC_, KW_, FC_*C_, 1, 1.f, 1, 3);
    gemm16<<<projBlocks,256,0,stream>>>(w2_i, hpad, fb2 + i*C_, yo, nullptr,
                                        FC_, C_, KW_, C_*FC_, 1, 1.f, 0, 0);
    ln_fuse<<<lnBlocks,256,0,stream>>>(xm, yo, g2 + i*C_, be2 + i*C_, xm, xt, 0);
  }

  writeout<<<(nE+255)/256,256,0,stream>>>(xm, mask, (float*)d_out);
}
